// CharRNN_4776003633158
// MI455X (gfx1250) — matrix operand order repaired and checked by reading, not compiled, not run on hardware
//
#include <hip/hip_runtime.h>

#define N_CHARS 256
#define EMB     1024
#define HID     2048
#define SEQ     8192
#define NWG     32      // persistent workgroups in the scan kernel
#define RWG     64      // hidden rows owned per workgroup (NWG*RWG == HID)

typedef __attribute__((ext_vector_type(16))) __bf16 bf16x16;
typedef __attribute__((ext_vector_type(8)))  float  f32x8;

union FragB16 {                 // one WMMA 16-bit operand: 8 VGPRs / 32 bytes
  bf16x16        v;
  unsigned short s[16];
  uint4          q[2];
};

__device__ __forceinline__ unsigned short f2bf(float f) {
  union { float f; unsigned u; } x; x.f = f;
  unsigned u = x.u + 0x7FFFu + ((x.u >> 16) & 1u);   // round-to-nearest-even
  return (unsigned short)(u >> 16);
}

// ---------------------------------------------------------------------------
// K0a: zero the per-step barrier counters and h0 (hist row 0). Runs every
// launch so the persistent-scan barrier state is deterministic & graph-safe.
// ---------------------------------------------------------------------------
__global__ void rnn_init_kernel(int* __restrict__ cnt, unsigned short* __restrict__ hist0) {
  const int i = blockIdx.x * blockDim.x + threadIdx.x;
  if (i < SEQ) cnt[i] = 0;
  if (i < HID) hist0[i] = 0;     // bf16 zero
}

// ---------------------------------------------------------------------------
// K0b: one-time f32 -> bf16 conversion (emb, wax, waa, wya). Runs once per
// launch; afterwards all hot loops read bf16 directly.
// ---------------------------------------------------------------------------
__global__ void cvt_bf16_kernel(const float* __restrict__ w,
                                unsigned short* __restrict__ o, int n) {
  const int i = (blockIdx.x * blockDim.x + threadIdx.x) * 8;
  if (i >= n) return;
  const float4 f0 = *(const float4*)(w + i);
  const float4 f1 = *(const float4*)(w + i + 4);
  ushort4 r0, r1;
  r0.x = f2bf(f0.x); r0.y = f2bf(f0.y); r0.z = f2bf(f0.z); r0.w = f2bf(f0.w);
  r1.x = f2bf(f1.x); r1.y = f2bf(f1.y); r1.z = f2bf(f1.z); r1.w = f2bf(f1.w);
  *(ushort4*)(o + i)     = r0;
  *(ushort4*)(o + i + 4) = r1;
}

// ---------------------------------------------------------------------------
// K1: xa[T][HID] = emb[seq[t]] @ wax_w^T + wax_b     (M=8192,K=1024,N=2048)
// One wave computes a 16x128 strip. Per K-step: issue A (2xb128) plus all
// 8 B fragments (16xb128) as one load clause, THEN run 8 independent WMMAs
// so the compiler can overlap tail loads with matrix ops (partial waits).
// ---------------------------------------------------------------------------
__global__ void __launch_bounds__(256) xa_gemm_kernel(
    const int* __restrict__ seq, const unsigned short* __restrict__ emb_bf,
    const unsigned short* __restrict__ wax_bf, const float* __restrict__ wax_b,
    float* __restrict__ xa) {
  const int wave  = blockIdx.x * 8 + (threadIdx.x >> 5);  // 8192 waves
  const int lane  = threadIdx.x & 31;
  const int mtile = wave >> 4;          // 512 M-tiles
  const int nblk  = wave & 15;          // 16 blocks of 128 cols
  const int m0 = mtile * 16, n0 = nblk * 128;
  const int lm = lane & 15, hi = lane >> 4, g = hi * 8;

  const unsigned short* arow = emb_bf + (size_t)seq[m0 + lm] * EMB;
  const unsigned short* wbase = wax_bf + (size_t)(n0 + lm) * EMB;

  f32x8 c[8];
#pragma unroll
  for (int j = 0; j < 8; ++j) c[j] = (f32x8)0.0f;

  for (int kk = 0; kk < EMB; kk += 32) {
    const int kb = kk + hi * 16;
    FragB16 a, b[8];
    a.q[0] = *(const uint4*)(arow + kb);            // K kb   .. +7
    a.q[1] = *(const uint4*)(arow + kb + 8);        // K kb+8 .. +15
#pragma unroll
    for (int j = 0; j < 8; ++j) {
      const unsigned short* wrow = wbase + (size_t)j * 16 * EMB;
      b[j].q[0] = *(const uint4*)(wrow + kb);       // K kb   .. +7
      b[j].q[1] = *(const uint4*)(wrow + kb + 8);   // K kb+8 .. +15
    }
#pragma unroll
    for (int j = 0; j < 8; ++j)
      c[j] = __builtin_amdgcn_wmma_f32_16x16x32_bf16(false, a.v, false, b[j].v,
                                                     (short)0, c[j], false, false);
  }
#pragma unroll
  for (int j = 0; j < 8; ++j) {
    const int col = n0 + j * 16 + lm;
    const float bias = wax_b[col];
#pragma unroll
    for (int r = 0; r < 8; ++r) {
      const int m = m0 + r + hi * 8;            // C layout: VGPR r, half hi
      xa[(size_t)m * HID + col] = c[j][r] + bias;
    }
  }
}

// ---------------------------------------------------------------------------
// K2: persistent serial scan. 32 WGs x 128 threads; wave w owns 16 rows of
// waa (bf16, streamed from L2 each step: 8 MB/step chip-wide, L2-resident).
// h_t broadcast via 4 KB LDS stage; GEMV as WMMA with B = h replicated into
// all 16 columns. K-loop unrolled x4 with batched loads and two independent
// accumulators so WMMAs pipeline instead of chaining D->C.
// ---------------------------------------------------------------------------
__global__ void __launch_bounds__(128) rnn_scan_kernel(
    const unsigned short* __restrict__ waa,   // [HID][HID] bf16
    const float* __restrict__ xa,             // [SEQ][HID] f32
    unsigned short* __restrict__ hist,        // [SEQ+1][HID] bf16 (row 0 = h0)
    int* __restrict__ cnt,                    // [SEQ] barrier counters (zeroed)
    float* __restrict__ h_final) {            // d_out tail [HID] f32
  __shared__ unsigned short h_lds[HID];       // 4 KB
  const int wave = threadIdx.x >> 5;          // 0..3
  const int lane = threadIdx.x & 31;
  const int lm = lane & 15, hi = lane >> 4, g = hi * 8;
  const int r0 = blockIdx.x * RWG + wave * 16;

  const unsigned short* arow = waa + (size_t)(r0 + lm) * HID;

  for (int t = 0; t < SEQ; ++t) {
    // Stage h_t into LDS with device-scope loads (must not hit stale L0:
    // row t was written by other WGPs at step t-1).
    {
      unsigned int* src = (unsigned int*)(hist + (size_t)t * HID);
      unsigned int* dst = (unsigned int*)h_lds;
      for (int i = threadIdx.x; i < HID / 2; i += 128)
        dst[i] = __hip_atomic_load(&src[i], __ATOMIC_RELAXED, __HIP_MEMORY_SCOPE_AGENT);
    }
    __syncthreads();

    f32x8 c0 = (f32x8)0.0f, c1 = (f32x8)0.0f;
    const uint4* hq = (const uint4*)h_lds;    // 8 bf16 per uint4
    for (int kk = 0; kk < HID; kk += 128) {
      FragB16 a[4], b[4];
#pragma unroll
      for (int u = 0; u < 4; ++u) {           // 8 global b128 in one clause
        const int k = kk + u * 32;
        a[u].q[0] = *(const uint4*)(arow + k + g);
        a[u].q[1] = *(const uint4*)(arow + k + 16 + g);
      }
#pragma unroll
      for (int u = 0; u < 4; ++u) {           // 8 LDS b128 reads
        const int qi = ((kk + u * 32) >> 3) + hi;
        b[u].q[0] = hq[qi];
        b[u].q[1] = hq[qi + 2];
      }
      c0 = __builtin_amdgcn_wmma_f32_16x16x32_bf16(false, a[0].v, false, b[0].v,
                                                   (short)0, c0, false, false);
      c1 = __builtin_amdgcn_wmma_f32_16x16x32_bf16(false, a[1].v, false, b[1].v,
                                                   (short)0, c1, false, false);
      c0 = __builtin_amdgcn_wmma_f32_16x16x32_bf16(false, a[2].v, false, b[2].v,
                                                   (short)0, c0, false, false);
      c1 = __builtin_amdgcn_wmma_f32_16x16x32_bf16(false, a[3].v, false, b[3].v,
                                                   (short)0, c1, false, false);
    }
    const f32x8 c = c0 + c1;

    // All 16 columns identical -> lanes 0/16 publish the 16 new h values.
    if (lm == 0) {
#pragma unroll
      for (int r = 0; r < 8; ++r) {
        const int row = r0 + r + hi * 8;
        const float v = tanhf(xa[(size_t)t * HID + row] + c[r]);
        hist[(size_t)(t + 1) * HID + row] = f2bf(v);
        if (t == SEQ - 1) h_final[row] = v;
      }
    }
    __threadfence();          // make this wave's h stores L2-visible
    __syncthreads();          // all waves in WG fenced before arrive
    if (threadIdx.x == 0) {
      __hip_atomic_fetch_add(&cnt[t], 1, __ATOMIC_RELEASE, __HIP_MEMORY_SCOPE_AGENT);
      while (__hip_atomic_load(&cnt[t], __ATOMIC_ACQUIRE, __HIP_MEMORY_SCOPE_AGENT) < NWG)
        __builtin_amdgcn_s_sleep(8);
    }
    __syncthreads();
  }
}

// ---------------------------------------------------------------------------
// K3: out[T][256] = h_hist @ wya_w^T + wya_b   (M=8192,K=2048,N=256)
// Same load-phase / compute-phase structure as K1.
// ---------------------------------------------------------------------------
__global__ void __launch_bounds__(256) out_gemm_kernel(
    const unsigned short* __restrict__ hist,  // [SEQ+1][HID] bf16
    const unsigned short* __restrict__ wya_bf, const float* __restrict__ wya_b,
    float* __restrict__ out) {
  const int wave  = blockIdx.x * 8 + (threadIdx.x >> 5);  // 1024 waves
  const int lane  = threadIdx.x & 31;
  const int mtile = wave >> 1;          // 512 M-tiles
  const int nblk  = wave & 1;           // 2 blocks of 128 cols
  const int m0 = mtile * 16, n0 = nblk * 128;
  const int lm = lane & 15, hi = lane >> 4, g = hi * 8;

  // out row t uses h after step t == hist row t+1
  const unsigned short* arow = hist + (size_t)(m0 + lm + 1) * HID;
  const unsigned short* wbase = wya_bf + (size_t)(n0 + lm) * HID;

  f32x8 c[8];
#pragma unroll
  for (int j = 0; j < 8; ++j) c[j] = (f32x8)0.0f;

  for (int kk = 0; kk < HID; kk += 32) {
    const int kb = kk + hi * 16;
    FragB16 a, b[8];
    a.q[0] = *(const uint4*)(arow + kb);
    a.q[1] = *(const uint4*)(arow + kb + 8);
#pragma unroll
    for (int j = 0; j < 8; ++j) {
      const unsigned short* wrow = wbase + (size_t)j * 16 * HID;
      b[j].q[0] = *(const uint4*)(wrow + kb);
      b[j].q[1] = *(const uint4*)(wrow + kb + 8);
    }
#pragma unroll
    for (int j = 0; j < 8; ++j)
      c[j] = __builtin_amdgcn_wmma_f32_16x16x32_bf16(false, a.v, false, b[j].v,
                                                     (short)0, c[j], false, false);
  }
#pragma unroll
  for (int j = 0; j < 8; ++j) {
    const int col = n0 + j * 16 + lm;
    const float bias = wya_b[col];
#pragma unroll
    for (int r = 0; r < 8; ++r) {
      const int m = m0 + r + hi * 8;
      out[(size_t)m * N_CHARS + col] = c[j][r] + bias;
    }
  }
}

// ---------------------------------------------------------------------------
extern "C" void kernel_launch(void* const* d_in, const int* in_sizes, int n_in,
                              void* d_out, int out_size, void* d_ws, size_t ws_size,
                              hipStream_t stream) {
  const int*   seq   = (const int*)  d_in[0];
  const float* emb   = (const float*)d_in[1];
  const float* wax_w = (const float*)d_in[2];
  const float* wax_b = (const float*)d_in[3];
  const float* waa_w = (const float*)d_in[4];
  const float* wya_w = (const float*)d_in[5];
  const float* wya_b = (const float*)d_in[6];
  float* out = (float*)d_out;

  // Workspace layout (~110 MB):
  char* ws = (char*)d_ws;
  size_t off = 0;
  unsigned short* waa_bf = (unsigned short*)(ws + off); off += (size_t)HID * HID * 2;        //  8 MB
  unsigned short* wax_bf = (unsigned short*)(ws + off); off += (size_t)HID * EMB * 2;        //  4 MB
  unsigned short* wya_bf = (unsigned short*)(ws + off); off += (size_t)N_CHARS * HID * 2;    //  1 MB
  unsigned short* emb_bf = (unsigned short*)(ws + off); off += (size_t)N_CHARS * EMB * 2;    // .5 MB
  float*          xa     = (float*)         (ws + off); off += (size_t)SEQ * HID * 4;        // 64 MB
  unsigned short* hist   = (unsigned short*)(ws + off); off += (size_t)(SEQ + 1) * HID * 2;  // 32 MB
  int*            cnt    = (int*)           (ws + off);                                      // 32 KB

  rnn_init_kernel<<<SEQ / 256, 256, 0, stream>>>(cnt, hist);

  cvt_bf16_kernel<<<(HID * HID)     / (8 * 256), 256, 0, stream>>>(waa_w, waa_bf, HID * HID);
  cvt_bf16_kernel<<<(HID * EMB)     / (8 * 256), 256, 0, stream>>>(wax_w, wax_bf, HID * EMB);
  cvt_bf16_kernel<<<(N_CHARS * HID) / (8 * 256), 256, 0, stream>>>(wya_w, wya_bf, N_CHARS * HID);
  cvt_bf16_kernel<<<(N_CHARS * EMB) / (8 * 256), 256, 0, stream>>>(emb,   emb_bf, N_CHARS * EMB);

  xa_gemm_kernel<<<1024, 256, 0, stream>>>(seq, emb_bf, wax_bf, wax_b, xa);
  rnn_scan_kernel<<<NWG, 128, 0, stream>>>(waa_bf, xa, hist, cnt,
                                           out + (size_t)SEQ * N_CHARS);
  out_gemm_kernel<<<128, 256, 0, stream>>>(hist, wya_bf, wya_b, out);
}
